// RNNTJointNetwork_25898652795604
// MI455X (gfx1250) — matrix operand order repaired and checked by reading, not compiled, not run on hardware
//
#include <hip/hip_runtime.h>
#include <hip/hip_bf16.h>

typedef __attribute__((ext_vector_type(16))) _Float16 v16h;
typedef __attribute__((ext_vector_type(8)))  _Float16 v8h;
typedef __attribute__((ext_vector_type(8)))  float    v8f;
typedef __attribute__((ext_vector_type(4)))  float    v4f;

union ABFrag { v16h v; v8h h[2]; };

#define ENC_DIM   512
#define DEC_DIM   640
#define JDIM      512
#define VOCAB     1024
#define BB        4
#define TT        256
#define UU        128

#define JPAD      8                 // 16-byte row pad: breaks power-of-2 LDS bank stride
#define JROW      (JDIM + JPAD)     // 520 halfs = 1040 B = 260 dwords (= 4 mod 64 banks)

// ---------------------------------------------------------------------------
// Kernel 1: W_out (J=512 x V=1024, f32 row-major) -> Wt (V x J, f16)
// ---------------------------------------------------------------------------
__global__ __launch_bounds__(256) void convert_wout_tr(
    const float* __restrict__ W, _Float16* __restrict__ Wt)
{
    const int tid = blockIdx.x * 256 + threadIdx.x;   // 0 .. V*J-1
    const int v = tid & (VOCAB - 1);
    const int j = tid >> 10;
    Wt[(size_t)v * JDIM + j] = (_Float16)W[tid];
}

// ---------------------------------------------------------------------------
// Kernel 2/3: C[M,N] = A[M,K] @ W[K,N] + bias (f32 in memory), f16 WMMA.
// 256 threads = 8 waves (4M x 2N), tile 64x64, K staged 32 at a time.
// ---------------------------------------------------------------------------
__global__ __launch_bounds__(256) void proj64x64_f16wmma(
    const float* __restrict__ A, const float* __restrict__ W,
    const float* __restrict__ bias, float* __restrict__ C,
    int M, int N, int K)
{
    __shared__ __align__(16) _Float16 ldsA[64][32 + JPAD];   // [m][k], 80B row stride
    __shared__ __align__(16) _Float16 ldsB[64][32 + JPAD];   // [n][k] (transposed W)

    const int tid  = threadIdx.x;
    const int lane = tid & 31;
    const int wave = tid >> 5;
    const int wm   = wave & 3;        // 0..3 -> M subtile
    const int wn   = wave >> 2;       // 0..1 -> N half
    const int m0   = blockIdx.y * 64;
    const int n0   = blockIdx.x * 64;

    v8f acc0 = {}; v8f acc1 = {};

    const int sr  = tid >> 2;          // 0..63  A stage row
    const int scg = (tid & 3) << 3;    // 0,8,16,24
    const int sn  = tid & 63;          // 0..63  B stage n
    const int skg = (tid >> 6) << 3;   // 0,8,16,24

    for (int k0 = 0; k0 < K; k0 += 32) {
        // Stage A (f32 -> f16), row-major in K
        {
            const float* src = A + (size_t)(m0 + sr) * K + k0 + scg;
            v4f a0 = *(const v4f*)src;
            v4f a1 = *(const v4f*)(src + 4);
            v8h h;
            h[0] = (_Float16)a0.x; h[1] = (_Float16)a0.y;
            h[2] = (_Float16)a0.z; h[3] = (_Float16)a0.w;
            h[4] = (_Float16)a1.x; h[5] = (_Float16)a1.y;
            h[6] = (_Float16)a1.z; h[7] = (_Float16)a1.w;
            *(v8h*)&ldsA[sr][scg] = h;
        }
        // Stage B transposed: ldsB[n][k] = W[k0+k][n0+n] (reads coalesced in n)
        {
            v8h h;
            #pragma unroll
            for (int j = 0; j < 8; ++j)
                h[j] = (_Float16)W[(size_t)(k0 + skg + j) * N + n0 + sn];
            *(v8h*)&ldsB[sn][skg] = h;
        }
        __syncthreads();

        ABFrag a, b0, b1;
        const int mrow = wm * 16 + (lane & 15);
        const int hi   = lane >> 4;
        a.h[0] = *(const v8h*)&ldsA[mrow][hi * 8];
        a.h[1] = *(const v8h*)&ldsA[mrow][16 + hi * 8];
        const int nb = wn * 32 + (lane & 15);
        b0.h[0] = *(const v8h*)&ldsB[nb][hi * 8];
        b0.h[1] = *(const v8h*)&ldsB[nb][16 + hi * 8];
        b1.h[0] = *(const v8h*)&ldsB[nb + 16][hi * 8];
        b1.h[1] = *(const v8h*)&ldsB[nb + 16][16 + hi * 8];

        acc0 = __builtin_amdgcn_wmma_f32_16x16x32_f16(
            false, a.v, false, b0.v, (short)0, acc0, false, false);
        acc1 = __builtin_amdgcn_wmma_f32_16x16x32_f16(
            false, a.v, false, b1.v, (short)0, acc1, false, false);
        __syncthreads();
    }

    const int hi    = lane >> 4;
    const int rbase = m0 + wm * 16 + hi * 8;
    const int c     = n0 + wn * 32 + (lane & 15);
    const float bo0 = bias[c];
    const float bo1 = bias[c + 16];
    #pragma unroll
    for (int r = 0; r < 8; ++r) {
        C[(size_t)(rbase + r) * N + c]      = acc0[r] + bo0;
        C[(size_t)(rbase + r) * N + c + 16] = acc1[r] + bo1;
    }
}

// ---------------------------------------------------------------------------
// Kernel 4 (dominant): fused joint(tanh) + output GEMM.
// One block owns 64 joint rows x the WHOLE vocab (1024 cols), so the tanh
// phase runs exactly once per row (67M total, the minimum).
// Phase 1: ldsJ[64][520] f16 = tanh(encP + decP)   (~65 KB LDS, padded rows)
// Phase 2: loop over 8 column chunks of 128; 8 waves as 2(M) x 4(N); each
// wave owns a 32x32 output per chunk (2x2 WMMA subtiles, 4 accumulators) so
// every A/B fragment feeds two WMMAs (2 b128 loads per WMMA).  B comes from
// the L2-resident f16 Wt (1 MB); next chunk is prefetched during compute.
// ---------------------------------------------------------------------------
__global__ __launch_bounds__(256) void joint_vocab_f16wmma(
    const float* __restrict__ encP, const float* __restrict__ decP,
    const _Float16* __restrict__ Wt, const float* __restrict__ bout,
    float* __restrict__ out)
{
    __shared__ __align__(16) _Float16 ldsJ[64][JROW];

    const int tid = threadIdx.x;
    const int m0  = blockIdx.x * 64;

    const int lane = tid & 31;
    const int wave = tid >> 5;
    const int wm   = wave & 1;        // 0..1 -> 32-row half
    const int wn   = wave >> 1;       // 0..3 -> 32-col group within chunk
    const int hi   = lane >> 4;
    const int l15  = lane & 15;

    const int ncsub = wn * 32 + l15;                       // col within chunk
    const _Float16* wbase0 = Wt + (size_t)ncsub * JDIM + hi * 8;
    const _Float16* wbase1 = wbase0 + (size_t)16 * JDIM;

    // Warm caches with chunk 0's Wt rows while phase 1 does transcendentals.
    #pragma unroll
    for (int pb = 0; pb < 1024; pb += 128) {
        __builtin_prefetch((const char*)wbase0 + pb, 0, 0);
        __builtin_prefetch((const char*)wbase1 + pb, 0, 0);
    }

    // ---- Phase 1: joint rows -> LDS (f16), once per row ----
    {
        const int r  = tid >> 2;              // 0..63 row in tile
        const int cg = (tid & 3) << 3;        // 0,8,16,24
        const int g  = m0 + r;                // global joint row
        const int encRow = g >> 7;            // / U1      -> b*T + t
        const int b      = g >> 15;           // / (T*U1)
        const int decRow = (b << 7) + (g & 127);
        const float* ep = encP + (size_t)encRow * JDIM;
        const float* dp = decP + (size_t)decRow * JDIM;
        #pragma unroll 4
        for (int c0 = cg; c0 < JDIM; c0 += 32) {
            v4f e0 = *(const v4f*)(ep + c0);
            v4f e1 = *(const v4f*)(ep + c0 + 4);
            v4f d0 = *(const v4f*)(dp + c0);
            v4f d1 = *(const v4f*)(dp + c0 + 4);
            v8h h;
            h[0] = (_Float16)tanhf(e0.x + d0.x);
            h[1] = (_Float16)tanhf(e0.y + d0.y);
            h[2] = (_Float16)tanhf(e0.z + d0.z);
            h[3] = (_Float16)tanhf(e0.w + d0.w);
            h[4] = (_Float16)tanhf(e1.x + d1.x);
            h[5] = (_Float16)tanhf(e1.y + d1.y);
            h[6] = (_Float16)tanhf(e1.z + d1.z);
            h[7] = (_Float16)tanhf(e1.w + d1.w);
            *(v8h*)&ldsJ[r][c0] = h;
        }
    }
    __syncthreads();

    // ---- Phase 2: loop the whole vocab in chunks of 128 columns ----
    const int mr0 = wm * 32 + l15;       // A subtile row 0
    const int mr1 = mr0 + 16;            // A subtile row 1
    const int rb0 = m0 + wm * 32 + hi * 8;
    const int rb1 = rb0 + 16;

    #pragma unroll 1
    for (int nc = 0; nc < VOCAB / 128; ++nc) {
        const int ncol = nc * 128 + ncsub;
        const _Float16* w0 = wbase0 + (size_t)(nc * 128) * JDIM;
        const _Float16* w1 = w0 + (size_t)16 * JDIM;

        // Prefetch next chunk's rows while this chunk computes.
        if (nc + 1 < VOCAB / 128) {
            const _Float16* p0 = w0 + (size_t)128 * JDIM;
            const _Float16* p1 = w1 + (size_t)128 * JDIM;
            #pragma unroll
            for (int pb = 0; pb < 1024; pb += 256) {
                __builtin_prefetch((const char*)p0 + pb, 0, 0);
                __builtin_prefetch((const char*)p1 + pb, 0, 0);
            }
        }

        v8f acc00 = {}, acc01 = {}, acc10 = {}, acc11 = {};
        #pragma unroll
        for (int ks = 0; ks < JDIM / 32; ++ks) {
            ABFrag a0, a1, b0, b1;
            a0.h[0] = *(const v8h*)&ldsJ[mr0][ks * 32 + hi * 8];
            a0.h[1] = *(const v8h*)&ldsJ[mr0][ks * 32 + 16 + hi * 8];
            a1.h[0] = *(const v8h*)&ldsJ[mr1][ks * 32 + hi * 8];
            a1.h[1] = *(const v8h*)&ldsJ[mr1][ks * 32 + 16 + hi * 8];
            b0.h[0] = *(const v8h*)(w0 + ks * 32);
            b0.h[1] = *(const v8h*)(w0 + ks * 32 + 16);
            b1.h[0] = *(const v8h*)(w1 + ks * 32);
            b1.h[1] = *(const v8h*)(w1 + ks * 32 + 16);
            acc00 = __builtin_amdgcn_wmma_f32_16x16x32_f16(
                false, a0.v, false, b0.v, (short)0, acc00, false, false);
            acc01 = __builtin_amdgcn_wmma_f32_16x16x32_f16(
                false, a0.v, false, b1.v, (short)0, acc01, false, false);
            acc10 = __builtin_amdgcn_wmma_f32_16x16x32_f16(
                false, a1.v, false, b0.v, (short)0, acc10, false, false);
            acc11 = __builtin_amdgcn_wmma_f32_16x16x32_f16(
                false, a1.v, false, b1.v, (short)0, acc11, false, false);
        }

        // Chunk epilogue: + b_out, f32 store
        const float bo0 = bout[ncol];
        const float bo1 = bout[ncol + 16];
        float* o0 = out + (size_t)rb0 * VOCAB + ncol;
        float* o1 = out + (size_t)rb1 * VOCAB + ncol;
        #pragma unroll
        for (int r = 0; r < 8; ++r) {
            o0[(size_t)r * VOCAB]      = acc00[r] + bo0;
            o0[(size_t)r * VOCAB + 16] = acc01[r] + bo1;
            o1[(size_t)r * VOCAB]      = acc10[r] + bo0;
            o1[(size_t)r * VOCAB + 16] = acc11[r] + bo1;
        }
    }
}

// ---------------------------------------------------------------------------
extern "C" void kernel_launch(void* const* d_in, const int* in_sizes, int n_in,
                              void* d_out, int out_size, void* d_ws, size_t ws_size,
                              hipStream_t stream)
{
    (void)in_sizes; (void)n_in; (void)out_size; (void)ws_size;

    const float* encoder_out = (const float*)d_in[0];
    const float* decoder_out = (const float*)d_in[1];
    const float* W_enc       = (const float*)d_in[2];
    const float* b_enc       = (const float*)d_in[3];
    const float* W_dec       = (const float*)d_in[4];
    const float* b_dec       = (const float*)d_in[5];
    const float* W_out       = (const float*)d_in[6];
    const float* b_out       = (const float*)d_in[7];
    float* out = (float*)d_out;

    // Workspace layout (4 MB total)
    char* ws = (char*)d_ws;
    float*    encP = (float*)ws;                             // 1024 x 512 f32 (2 MB)
    float*    decP = (float*)(ws + ((size_t)2 << 20));       //  512 x 512 f32 (1 MB)
    _Float16* Wt   = (_Float16*)(ws + ((size_t)3 << 20));    // 1024 x 512 f16 (1 MB)

    // 1) W_out -> transposed f16
    convert_wout_tr<<<(VOCAB * JDIM) / 256, 256, 0, stream>>>(W_out, Wt);

    // 2) enc projection: (B*T=1024) x 512, K=512
    proj64x64_f16wmma<<<dim3(JDIM / 64, (BB * TT) / 64), 256, 0, stream>>>(
        encoder_out, W_enc, b_enc, encP, BB * TT, JDIM, ENC_DIM);

    // 3) dec projection: (B*U1=512) x 512, K=640
    proj64x64_f16wmma<<<dim3(JDIM / 64, (BB * UU) / 64), 256, 0, stream>>>(
        decoder_out, W_dec, b_dec, decP, BB * UU, JDIM, DEC_DIM);

    // 4) fused tanh-joint + output GEMM: 131072 x 1024 (K=512), full vocab per block
    joint_vocab_f16wmma<<<(BB * TT * UU) / 64, 256, 0, stream>>>(
        encP, decP, Wt, b_out, out);
}
